// RPEMultiHeadAttention_2791728742878
// MI455X (gfx1250) — matrix operand order repaired and checked by reading, not compiled, not run on hardware
//
#include <hip/hip_runtime.h>
#include <hip/hip_bf16.h>

typedef __attribute__((ext_vector_type(16))) _Float16 v16h;
typedef __attribute__((ext_vector_type(8)))  float    v8f;
typedef __attribute__((ext_vector_type(2)))  _Float16 h2;
typedef __attribute__((ext_vector_type(4)))  _Float16 h4;
typedef __attribute__((ext_vector_type(4)))  unsigned int u32x4;
typedef __attribute__((ext_vector_type(8)))  unsigned int u32x8;

#define N_  2048
#define M_  2048
#define D_  256
#define H_  8
#define DH_ 32
#define INV_SQRT_DH 0.17677669529663687f

// A-fragment K order this code uses for V_WMMA_F32_16X16X32_F16 (contiguous k = 16h + s; B matches):
// lanes 0-15: VGPR p holds K = {2p,2p+1}
// lanes 16-31: same +16
__device__ __forceinline__ int aKmap(int p, int hi) {
  return 2 * p + hi * 16;
}

// ---------------------------------------------------------------------------
// Kernel 1: fused QKV projection, f32 -> f16, head-major outputs.
//   Qh[h][n][dh], Kh[h][m][dh]  (row-major per head, DH contiguous)
//   Vt[h][dh][m]                (transposed: m contiguous, for attn@V B-frags
//                                and a dense 2D TDM descriptor per head)
// One wave computes one 16x16 output tile via 8 K=32 WMMA steps.
// ---------------------------------------------------------------------------
__global__ __launch_bounds__(256) void qkv_proj_kernel(
    const float* __restrict__ xq, const float* __restrict__ xk, const float* __restrict__ xv,
    const float* __restrict__ Wq, const float* __restrict__ Wk, const float* __restrict__ Wv,
    const float* __restrict__ bq, const float* __restrict__ bk, const float* __restrict__ bv,
    _Float16* __restrict__ Qh, _Float16* __restrict__ Kh, _Float16* __restrict__ Vt)
{
  const int wave = blockIdx.x * (blockDim.x >> 5) + (threadIdx.x >> 5);
  const int lane = threadIdx.x & 31;
  const int hi   = lane >> 4;
  const int lcol = lane & 15;

  const int which = wave / (128 * 16);       // 0=q 1=k 2=v
  const int rem   = wave % (128 * 16);
  const int n0    = (rem >> 4) << 4;         // token-tile base
  const int o0    = (rem & 15) << 4;         // out-feature-tile base

  const float* X  = which == 0 ? xq : (which == 1 ? xk : xv);
  const float* W  = which == 0 ? Wq : (which == 1 ? Wk : Wv);
  const float* Bv = which == 0 ? bq : (which == 1 ? bk : bv);

  const int row  = n0 + lcol;   // A-matrix row (token)
  const int ocol = o0 + lcol;   // B-matrix column (output feature)

  v8f acc = {};
  #pragma unroll
  for (int kt = 0; kt < 8; ++kt) {
    const int k0 = kt * 32;
    v16h a, b;
    const float* arow = X + row * D_ + k0;
    #pragma unroll
    for (int p = 0; p < 8; ++p) {
      float2 f = *(const float2*)(arow + aKmap(p, hi));
      a[2 * p]     = (_Float16)f.x;
      a[2 * p + 1] = (_Float16)f.y;
    }
    // out = x @ W^T  =>  B[k][o] = W[o][k]; B lane holds col o, K = e + 16*hi
    const float* wrow = W + ocol * D_ + k0 + hi * 16;
    #pragma unroll
    for (int e = 0; e < 16; e += 2) {
      float2 f = *(const float2*)(wrow + e);
      b[e]     = (_Float16)f.x;
      b[e + 1] = (_Float16)f.y;
    }
    acc = __builtin_amdgcn_wmma_f32_16x16x32_f16(false, a, false, b, (short)0, acc,
                                                 false, false);
  }

  const float bias = Bv[ocol];
  const int h  = ocol >> 5;   // ocol / DH
  const int dh = ocol & 31;

  if (which == 2) {           // V: store transposed
    #pragma unroll
    for (int r = 0; r < 8; ++r) {
      const int tok = n0 + r + hi * 8;
      Vt[(h * DH_ + dh) * M_ + tok] = (_Float16)(acc[r] + bias);
    }
  } else {
    _Float16* dst = (which == 0) ? Qh : Kh;
    #pragma unroll
    for (int r = 0; r < 8; ++r) {
      const int tok = n0 + r + hi * 8;
      dst[(h * N_ + tok) * DH_ + dh] = (_Float16)(acc[r] + bias);
    }
  }
}

// ---------------------------------------------------------------------------
// Kernel 2: head-shared RPE bias  p[i][j] = dot(embed_qk[i][j][:], Wp) + bp
// Computed once (not per head). 4 outputs / thread, 16B-aligned loads.
// ---------------------------------------------------------------------------
__global__ __launch_bounds__(256) void rpe_bias_kernel(
    const float* __restrict__ e3, const float* __restrict__ Wp,
    const float* __restrict__ bp, float* __restrict__ P)
{
  const int t = blockIdx.x * blockDim.x + threadIdx.x;   // one per 4 (i,j) pairs
  const float w0 = Wp[0], w1 = Wp[1], w2 = Wp[2], b = bp[0];
  const int base = t * 4;
  const float* src = e3 + (size_t)base * 3;              // 48B-aligned
  float4 x0 = *(const float4*)(src);
  float4 x1 = *(const float4*)(src + 4);
  float4 x2 = *(const float4*)(src + 8);
  float4 o;
  o.x = x0.x * w0 + x0.y * w1 + x0.z * w2 + b;
  o.y = x0.w * w0 + x1.x * w1 + x1.y * w2 + b;
  o.z = x1.z * w0 + x1.w * w1 + x2.x * w2 + b;
  o.w = x2.y * w0 + x2.z * w1 + x2.w * w2 + b;
  *(float4*)(P + base) = o;
}

// ---------------------------------------------------------------------------
// Kernel 3: attention. One 256-thread (8-wave) workgroup per (head, 16-query
// tile). LDS: 16x2048 f32 score panel (128 KB) + 16x2048 f16 prob panel
// (64 KB) + 2 KB reduction = 194 KB of the 320 KB/WGP. Phases:
//   1. 8 waves x 16 chunks: S = WMMA(Q,K^T) -> scale/bias/mask -> LDS
//   2. exact softmax (shfl_xor row reductions), write attn + f16 probs
//   2b. score panel is dead -> TDM DMA of the 128 KB per-head V panel into
//       that LDS region (tensor_load_to_lds + s_wait_tensorcnt)
//   3. 8 waves split K=2048: hidden += WMMA(P, V_lds), ds_add_f32 reduction
// ---------------------------------------------------------------------------
__global__ __launch_bounds__(256) void attn_kernel(
    const _Float16* __restrict__ Qh, const _Float16* __restrict__ Kh,
    const _Float16* __restrict__ Vt, const float* __restrict__ Pb,
    const float* __restrict__ AF, const float* __restrict__ kw,
    const unsigned char* __restrict__ km,
    float* __restrict__ hidden, float* __restrict__ attn)
{
  extern __shared__ char smem[];
  float*    Sbuf = (float*)smem;                                 // 16*2048*4 = 128 KB
  _Float16* Pbuf = (_Float16*)(smem + 16 * M_ * 4);              // 16*2048*2 =  64 KB
  float*    Hacc = (float*)(smem + 16 * M_ * 4 + 16 * M_ * 2);   // 16*32*4   =   2 KB

  const int wave = threadIdx.x >> 5;
  const int lane = threadIdx.x & 31;
  const int hi   = lane >> 4;
  const int lcol = lane & 15;
  const int h    = blockIdx.x & 7;     // heads innermost -> L2 reuse of Pb/AF rows
  const int n0   = (blockIdx.x >> 3) * 16;

  // zero hidden accumulator
  Hacc[threadIdx.x]       = 0.f;
  Hacc[threadIdx.x + 256] = 0.f;

  // Q fragment: DH=32 == one WMMA K-step
  v16h aq;
  {
    const _Float16* qrow = Qh + ((size_t)h * N_ + n0 + lcol) * DH_;
    #pragma unroll
    for (int p = 0; p < 8; ++p) {
      h2 v = *(const h2*)(qrow + aKmap(p, hi));
      aq[2 * p]     = v.x;
      aq[2 * p + 1] = v.y;
    }
  }

  // ---- Phase 1: scores into LDS -------------------------------------------
  for (int c = wave * 16; c < wave * 16 + 16; ++c) {
    const int m0 = c * 16;
    v16h bk;
    const _Float16* krow = Kh + ((size_t)h * M_ + m0 + lcol) * DH_ + hi * 16;
    __builtin_prefetch(krow + 16 * DH_, 0, 0);   // global_prefetch_b8 next chunk
    #pragma unroll
    for (int e = 0; e < 16; e += 2) {
      h2 v = *(const h2*)(krow + e);
      bk[e]     = v.x;
      bk[e + 1] = v.y;
    }
    v8f s = {};
    s = __builtin_amdgcn_wmma_f32_16x16x32_f16(false, aq, false, bk, (short)0, s,
                                               false, false);
    const int j = m0 + lcol;
    const float kwv   = kw[j];
    const bool masked = km[j] != 0;
    #pragma unroll
    for (int r = 0; r < 8; ++r) {
      const int il  = r + hi * 8;
      const int idx = (n0 + il) * M_ + j;
      float sv = (s[r] + Pb[idx]) * INV_SQRT_DH * AF[idx] * kwv;
      if (masked) sv = -1e30f;
      Sbuf[il * M_ + j] = sv;
    }
  }
  __syncthreads();

  // ---- Phase 2: exact softmax, write attn + f16 probs (4 cols/thread) -----
  {
    const int row = threadIdx.x >> 4;       // 16 threads per query row
    const int sub = threadIdx.x & 15;       // groups align with wave lane-halves
    float* srow = Sbuf + row * M_;
    float mx = -1e30f;
    for (int q = sub * 4; q < M_; q += 64) {
      float4 sv = *(const float4*)(srow + q);
      mx = fmaxf(mx, fmaxf(fmaxf(sv.x, sv.y), fmaxf(sv.z, sv.w)));
    }
    #pragma unroll
    for (int o = 8; o >= 1; o >>= 1) mx = fmaxf(mx, __shfl_xor(mx, o, 32));
    float sum = 0.f;
    for (int q = sub * 4; q < M_; q += 64) {           // cache exp() in place
      float4 sv = *(const float4*)(srow + q);
      sv.x = __expf(sv.x - mx); sv.y = __expf(sv.y - mx);
      sv.z = __expf(sv.z - mx); sv.w = __expf(sv.w - mx);
      *(float4*)(srow + q) = sv;
      sum += (sv.x + sv.y) + (sv.z + sv.w);
    }
    #pragma unroll
    for (int o = 8; o >= 1; o >>= 1) sum += __shfl_xor(sum, o, 32);
    const float inv = 1.f / sum;
    float* arow = attn + ((size_t)h * N_ + n0 + row) * M_;
    for (int q = sub * 4; q < M_; q += 64) {
      float4 ev = *(const float4*)(srow + q);
      float4 pv = make_float4(ev.x * inv, ev.y * inv, ev.z * inv, ev.w * inv);
      *(float4*)(arow + q) = pv;
      h4 ph = { (_Float16)pv.x, (_Float16)pv.y, (_Float16)pv.z, (_Float16)pv.w };
      *(h4*)(Pbuf + row * M_ + q) = ph;
    }
  }
  __syncthreads();   // all waves done with Sbuf (scores dead from here on)

  // ---- Phase 2b: TDM DMA the per-head V panel (32 x 2048 f16 = 128 KB)
  //      into the dead Sbuf region. 2D descriptor, 2-SGPR-group form
  //      (VADDR2/VADDR3 both NULL, legal for tensors up to 2D).
  if (wave == 0) {
    const unsigned lds_base = (unsigned)(uintptr_t)smem;   // flat[31:0] == LDS offset
    const unsigned long long ga =
        (unsigned long long)(uintptr_t)(Vt + (size_t)h * DH_ * M_);
    u32x4 g0;
    g0[0] = 1u;                                            // count = 1 valid D#
    g0[1] = lds_base;                                      // lds_addr (bytes)
    g0[2] = (unsigned)(ga & 0xFFFFFFFFu);                  // global_addr[31:0]
    g0[3] = (unsigned)((ga >> 32) & 0x01FFFFFFu) | (2u << 30);  // addr[56:32] | type=2
    u32x8 g1;
    g1[0] = 1u << 16;                 // data_size = 1 -> 2-byte elements
    g1[1] = (unsigned)M_ << 16;       // tensor_dim0 = 2048   (bits 63:48)
    g1[2] = (unsigned)DH_ << 16;      // tensor_dim1 = 32     (bits 95:80)
    g1[3] = (unsigned)M_ << 16;       // tile_dim0   = 2048   (bits 127:112)
    g1[4] = (unsigned)DH_;            // tile_dim1   = 32     (bits 143:128)
    g1[5] = (unsigned)M_;             // tensor_dim0_stride = 2048 (bits 191:160)
    g1[6] = 0u;
    g1[7] = 0u;
    asm volatile("tensor_load_to_lds %0, %1" :: "s"(g0), "s"(g1) : "memory");
    __builtin_amdgcn_s_wait_tensorcnt(0);
  }
  __syncthreads();   // release all waves once V panel is resident in LDS

  // ---- Phase 3: hidden = P @ V, K=2048 split 8 ways across waves ----------
  const _Float16* Vl = (const _Float16*)smem;   // V panel: Vl[dh][m], m contiguous
  v8f h0 = {}, h1 = {};
  for (int c = wave * 8; c < wave * 8 + 8; ++c) {
    const int kb = c * 32;
    v16h pa, b0, b1;
    const _Float16* prow = Pbuf + lcol * M_ + kb;       // LDS A-frag (ds_load)
    #pragma unroll
    for (int p = 0; p < 8; ++p) {
      h2 v = *(const h2*)(prow + aKmap(p, hi));
      pa[2 * p]     = v.x;
      pa[2 * p + 1] = v.y;
    }
    const _Float16* v0 = Vl + (lcol)      * M_ + kb + hi * 16;   // dh = lcol
    const _Float16* v1 = Vl + (16 + lcol) * M_ + kb + hi * 16;   // dh = 16+lcol
    #pragma unroll
    for (int e = 0; e < 16; e += 2) {
      h2 a0 = *(const h2*)(v0 + e); b0[e] = a0.x; b0[e + 1] = a0.y;
      h2 a1 = *(const h2*)(v1 + e); b1[e] = a1.x; b1[e + 1] = a1.y;
    }
    h0 = __builtin_amdgcn_wmma_f32_16x16x32_f16(false, pa, false, b0, (short)0, h0,
                                                false, false);
    h1 = __builtin_amdgcn_wmma_f32_16x16x32_f16(false, pa, false, b1, (short)0, h1,
                                                false, false);
  }
  // cross-wave reduce with LDS float atomics (ds_add_f32)
  #pragma unroll
  for (int r = 0; r < 8; ++r) {
    const int il = r + hi * 8;
    atomicAdd(&Hacc[il * DH_ + lcol],      h0[r]);
    atomicAdd(&Hacc[il * DH_ + 16 + lcol], h1[r]);
  }
  __syncthreads();

  #pragma unroll
  for (int t = 0; t < 2; ++t) {
    const int idx = threadIdx.x * 2 + t;
    const int il = idx >> 5, dh = idx & 31;
    hidden[(size_t)(n0 + il) * D_ + h * DH_ + dh] = Hacc[idx];
  }
}

// ---------------------------------------------------------------------------
extern "C" void kernel_launch(void* const* d_in, const int* in_sizes, int n_in,
                              void* d_out, int out_size, void* d_ws, size_t ws_size,
                              hipStream_t stream) {
  (void)in_sizes; (void)n_in; (void)out_size; (void)ws_size;

  const float* input_q  = (const float*)d_in[0];
  const float* input_k  = (const float*)d_in[1];
  const float* input_v  = (const float*)d_in[2];
  const float* embed_qk = (const float*)d_in[3];
  const float* key_w    = (const float*)d_in[4];
  const unsigned char* key_m = (const unsigned char*)d_in[5];  // jnp bool -> 1 byte
  const float* att_f    = (const float*)d_in[6];
  const float* Wq = (const float*)d_in[7];
  const float* bq = (const float*)d_in[8];
  const float* Wk = (const float*)d_in[9];
  const float* bk = (const float*)d_in[10];
  const float* Wv = (const float*)d_in[11];
  const float* bv = (const float*)d_in[12];
  const float* Wp = (const float*)d_in[13];
  const float* bp = (const float*)d_in[14];

  char* ws = (char*)d_ws;
  const size_t headBytes = (size_t)H_ * N_ * DH_ * sizeof(_Float16);  // 1 MB each
  _Float16* Qh = (_Float16*)(ws);
  _Float16* Kh = (_Float16*)(ws + headBytes);
  _Float16* Vt = (_Float16*)(ws + 2 * headBytes);
  float*    Pb = (float*)(ws + 3 * headBytes);                        // 16 MB

  float* hidden = (float*)d_out;
  float* attn   = hidden + (size_t)N_ * D_;

  // 1) QKV projections: 3 * 2048 tiles, 8 waves/block -> 768 blocks
  qkv_proj_kernel<<<768, 256, 0, stream>>>(input_q, input_k, input_v,
                                           Wq, Wk, Wv, bq, bk, bv, Qh, Kh, Vt);
  // 2) head-shared RPE bias (once, not per head)
  rpe_bias_kernel<<<(N_ * M_ / 4) / 256, 256, 0, stream>>>(embed_qk, Wp, bp, Pb);

  // 3) attention: 128 query tiles x 8 heads, 194 KB dynamic LDS (320 KB/WGP)
  const size_t smem = (size_t)16 * M_ * 4 + (size_t)16 * M_ * 2 + 16 * DH_ * 4;
  (void)hipFuncSetAttribute((const void*)attn_kernel,
                            hipFuncAttributeMaxDynamicSharedMemorySize, (int)smem);
  attn_kernel<<<(N_ / 16) * H_, 256, smem, stream>>>(Qh, Kh, Vt, Pb, att_f,
                                                     key_w, key_m, hidden, attn);
}
